// MultiHeadAttention_51196010169054
// MI455X (gfx1250) — matrix operand order repaired and checked by reading, not compiled, not run on hardware
//
#include <hip/hip_runtime.h>

typedef unsigned short u16;
typedef unsigned int u32;
typedef unsigned long long u64;
typedef __attribute__((ext_vector_type(16))) __bf16 v16bf;
typedef __attribute__((ext_vector_type(8)))  float  v8f;
typedef __attribute__((ext_vector_type(4)))  unsigned int u32x4;
typedef __attribute__((ext_vector_type(8)))  int i32x8;
typedef __attribute__((ext_vector_type(4)))  int i32x4;

#define SEQLEN 4096
#define DMODEL 1024
#define NHEADS 16
#define DHEAD  64

// Tensor Data Mover availability (this toolchain: clang-23 6-arg form)
#if defined(__has_builtin)
#if __has_builtin(__builtin_amdgcn_tensor_load_to_lds) && \
    __has_builtin(__builtin_amdgcn_s_wait_tensorcnt)
#define USE_TDM 1
#endif
#endif
#ifndef USE_TDM
#define USE_TDM 0
#endif

// ---------------------------------------------------------------- helpers

__device__ __forceinline__ u16 f2bf_rne(float f) {
  u32 x = __float_as_uint(f);
  x += 0x7fffu + ((x >> 16) & 1u);   // round-to-nearest-even
  return (u16)(x >> 16);
}

union FragU { v16bf v; uint4 q[2]; };

// A-matrix fragment (16x32, bf16), src row-major, ld multiple of 8.
// ISA layout: lanes 0-15 = rows 0-15 holding K {0..7,16..23};
//             lanes 16-31 = rows 0-15 holding K {8..15,24..31}.
__device__ __forceinline__ v16bf load_frag_a(const u16* p, int ld) {
  const int lane = threadIdx.x & 31;
  const int row  = lane & 15;
  const int koff = (lane >> 4) << 3;          // 0 or 8
  const u16* s = p + row * ld + koff;
  FragU f;
  f.q[0] = *(const uint4*)(s);                // K koff..koff+7
  f.q[1] = *(const uint4*)(s + 16);           // K koff+16..koff+23
  return f.v;
}

// B-matrix fragment (32x16, bf16) from "Bmat" stored as N x K row-major
// (i.e. B[k][n] = Bmat[n][k]).  lanes 0-15 = col n, K {0..7,16..23};
// lanes 16-31 = col n, K {8..15,24..31}.  ld multiple of 8.
__device__ __forceinline__ v16bf load_frag_b(const u16* p, int ld) {
  const int lane = threadIdx.x & 31;
  const int col  = lane & 15;
  const int koff = (lane >> 4) << 3;          // 0 or 8
  const u16* s = p + col * ld + koff;
  FragU f;
  f.q[0] = *(const uint4*)(s);
  f.q[1] = *(const uint4*)(s + 16);
  return f.v;
}

__device__ __forceinline__ v8f wmma_bf16(v16bf a, v16bf b, v8f c) {
  return __builtin_amdgcn_wmma_f32_16x16x32_bf16(
      /*neg_a=*/false, a, /*neg_b=*/false, b,
      /*c_mod=*/(short)0, c, /*reuse_a=*/false, /*reuse_b=*/false);
}

// ---------------------------------------------------------------- convert

__global__ __launch_bounds__(256) void f32_to_bf16_kernel(
    const float* __restrict__ in, u16* __restrict__ out, int n) {
  int i = blockIdx.x * 256 + threadIdx.x;
  if (i < n) out[i] = f2bf_rne(in[i]);
}

// ---------------------------------------------------------------- GEMM C = A * B^T
// A: M x K bf16 row-major, B: N x K bf16 row-major, C: M x N (f32 or bf16)
// 128 threads / block; block tile 64x64; wave w handles rows 16w..16w+15.

template <bool F32OUT>
__global__ __launch_bounds__(128) void gemm_abt(
    const u16* __restrict__ A, const u16* __restrict__ B,
    void* __restrict__ Cv, int M, int N, int K) {
  const int wave = threadIdx.x >> 5;
  const int lane = threadIdx.x & 31;
  const int row0 = blockIdx.x * 64 + wave * 16;
  const int col0 = blockIdx.y * 64;

  const v8f vzero = {0.f, 0.f, 0.f, 0.f, 0.f, 0.f, 0.f, 0.f};
  v8f acc[4];
#pragma unroll
  for (int t = 0; t < 4; ++t) acc[t] = vzero;

  for (int k = 0; k < K; k += 32) {
    v16bf a = load_frag_a(A + (size_t)row0 * K + k, K);
#pragma unroll
    for (int t = 0; t < 4; ++t) {
      v16bf b = load_frag_b(B + (size_t)(col0 + t * 16) * K + k, K);
      acc[t] = wmma_bf16(a, b, acc[t]);
    }
  }

  const int hlf = lane >> 4;
  const int cn  = lane & 15;
#pragma unroll
  for (int t = 0; t < 4; ++t) {
#pragma unroll
    for (int r = 0; r < 8; ++r) {
      size_t off = (size_t)(row0 + r + 8 * hlf) * N + col0 + t * 16 + cn;
      if (F32OUT) ((float*)Cv)[off] = acc[t][r];
      else        ((u16*)Cv)[off]   = f2bf_rne(acc[t][r]);
    }
  }
}

// ---------------------------------------------------------------- flash attention
// grid = (SEQ/64, NHEADS), block = 128 (4 waves). Wave w: 16 query rows.
// Q/K/V: bf16 (SEQ, DMODEL), head h occupies cols h*64..h*64+63.
// Output O: bf16 (SEQ, DMODEL).

__global__ __launch_bounds__(128) void flash_attn_kernel(
    const u16* __restrict__ Q, const u16* __restrict__ K,
    const u16* __restrict__ V, u16* __restrict__ O) {
  __shared__ u16 sK[64 * 64];        // [key][d]
  __shared__ u16 sVt[64 * 64];       // [d][key]  (transposed)
  __shared__ u16 sP[4 * 16 * 64];    // per-wave P tile [row][key]

  const int qb   = blockIdx.x;
  const int h    = blockIdx.y;
  const int tid  = threadIdx.x;
  const int wave = tid >> 5;
  const int lane = tid & 31;
  const int hlf  = lane >> 4;
  const int cn   = lane & 15;
  const int row0 = qb * 64 + wave * 16;

  // Q fragments for this wave's 16 rows (d_head = 64 -> two K=32 blocks)
  const u16* qbase = Q + (size_t)row0 * DMODEL + h * DHEAD;
  v16bf aq0 = load_frag_a(qbase, DMODEL);
  v16bf aq1 = load_frag_a(qbase + 32, DMODEL);

  const v8f vzero = {0.f, 0.f, 0.f, 0.f, 0.f, 0.f, 0.f, 0.f};
  v8f o[4];
  float mi[8], li[8];
#pragma unroll
  for (int t = 0; t < 4; ++t) o[t] = vzero;
#pragma unroll
  for (int r = 0; r < 8; ++r) { mi[r] = -__builtin_inff(); li[r] = 0.f; }

  u16* sPw = sP + wave * (16 * 64);
  const float kLog2e = 1.44269504f;

#if USE_TDM
  // Constant part of the K-tile Tensor DMA descriptor (D#), §8 of the TDM ISA:
  // 64x64 bf16 tile, row stride DMODEL elements, dest = sK (contiguous).
  i32x8 tg1;
  tg1[0] = 0x10000;            // data_size = 1 (2 bytes/elem)
  tg1[1] = 64 << 16;           // tensor_dim0[15:0]  = 64   (bits 63:48)
  tg1[2] = 64 << 16;           // tensor_dim1[15:0]  = 64   (bits 95:80)
  tg1[3] = 64 << 16;           // tile_dim0 = 64            (bits 127:112)
  tg1[4] = 64;                 // tile_dim1 = 64, tile_dim2 = 0
  tg1[5] = DMODEL;             // tensor_dim0_stride = 1024 (bits 191:160)
  tg1[6] = 0;                  // stride hi, tensor_dim1_stride lo
  tg1[7] = 0;
  const i32x4 tgz4 = {0, 0, 0, 0};               // groups 2/3 unused (2-D)
  const i32x8 tgz8 = {0, 0, 0, 0, 0, 0, 0, 0};   // trailing group, zero-filled
  const u32 ldsK = (u32)(size_t)(void*)sK;  // flat LDS addr[31:0] = LDS offset
#endif

  for (int kb = 0; kb <= qb; ++kb) {
    __syncthreads();   // previous tile fully consumed before overwrite

#if USE_TDM
    // K tile via Tensor Data Mover: one DMA descriptor, issued by wave 0.
    if (wave == 0) {
      u64 ga = (u64)(size_t)(const void*)(K + (size_t)(kb * 64) * DMODEL + h * DHEAD);
      u32x4 g0;
      g0[0] = 1u;                              // count = 1, user mode
      g0[1] = ldsK;                            // lds_addr (bytes)
      g0[2] = (u32)ga;                         // global_addr[31:0]
      g0[3] = (u32)(ga >> 32) | (2u << 30);    // global_addr[56:32] | type=2
      __builtin_amdgcn_tensor_load_to_lds(g0, tg1, tgz4, tgz4, tgz8, 0);
      __builtin_amdgcn_s_wait_tensorcnt(0);
    }
#else
    // K tile: 4096 bf16, 8 per thread per pass, 4 passes, coalesced
#pragma unroll
    for (int e = tid * 8; e < 64 * 64; e += 128 * 8) {
      int kr = e >> 6, d = e & 63;
      *(uint4*)(sK + e) =
          *(const uint4*)(K + (size_t)(kb * 64 + kr) * DMODEL + h * DHEAD + d);
    }
#endif

    // V tile, transposed into sVt[d][key]
#pragma unroll
    for (int e = tid * 8; e < 64 * 64; e += 128 * 8) {
      int kr = e >> 6, d = e & 63;
      uint4 raw =
          *(const uint4*)(V + (size_t)(kb * 64 + kr) * DMODEL + h * DHEAD + d);
      const u16* w = (const u16*)&raw;
#pragma unroll
      for (int j = 0; j < 8; ++j) sVt[(d + j) * 64 + kr] = w[j];
    }
    // prefetch next K/V tile while we compute (global_prefetch_b8)
    if (kb < qb) {
      int pr = tid & 63;
      const u16* src = ((tid < 64) ? K : V) +
                       (size_t)((kb + 1) * 64 + pr) * DMODEL + h * DHEAD;
      __builtin_prefetch(src, 0, 0);
    }
    __syncthreads();

    // S = Q K^T  (16 x 64 per wave)
    v8f s[4];
#pragma unroll
    for (int t = 0; t < 4; ++t) {
      s[t] = vzero;
      v16bf b0 = load_frag_b(sK + (t * 16) * 64 + 0, 64);
      s[t] = wmma_bf16(aq0, b0, s[t]);
      v16bf b1 = load_frag_b(sK + (t * 16) * 64 + 32, 64);
      s[t] = wmma_bf16(aq1, b1, s[t]);
    }
#pragma unroll
    for (int t = 0; t < 4; ++t) s[t] *= 0.125f;   // 1/sqrt(64)

    // causal mask on diagonal block
    if (kb == qb) {
#pragma unroll
      for (int t = 0; t < 4; ++t) {
        int col = kb * 64 + t * 16 + cn;
#pragma unroll
        for (int r = 0; r < 8; ++r) {
          int row = row0 + r + 8 * hlf;
          if (col > row) s[t][r] = -__builtin_inff();
        }
      }
    }

    // online softmax. C-layout: element (M = r + 8*hlf, N = cn) in s[t][r];
    // row reductions across the 16 lanes of this half via xor-shuffles.
#pragma unroll
    for (int r = 0; r < 8; ++r) {
      float mx = fmaxf(fmaxf(s[0][r], s[1][r]), fmaxf(s[2][r], s[3][r]));
      mx = fmaxf(mx, __shfl_xor(mx, 1, 32));
      mx = fmaxf(mx, __shfl_xor(mx, 2, 32));
      mx = fmaxf(mx, __shfl_xor(mx, 4, 32));
      mx = fmaxf(mx, __shfl_xor(mx, 8, 32));
      float mnew = fmaxf(mi[r], mx);
      float corr = __builtin_exp2f((mi[r] - mnew) * kLog2e);
      float rsum = 0.f;
#pragma unroll
      for (int t = 0; t < 4; ++t) {
        float p = __builtin_exp2f((s[t][r] - mnew) * kLog2e);
        s[t][r] = p;
        rsum += p;
      }
      rsum += __shfl_xor(rsum, 1, 32);
      rsum += __shfl_xor(rsum, 2, 32);
      rsum += __shfl_xor(rsum, 4, 32);
      rsum += __shfl_xor(rsum, 8, 32);
      li[r] = li[r] * corr + rsum;
      mi[r] = mnew;
#pragma unroll
      for (int t = 0; t < 4; ++t) o[t][r] *= corr;
    }

    // P (C-layout) -> LDS -> A-layout for the PV matmul (wave-private tile)
#pragma unroll
    for (int t = 0; t < 4; ++t)
#pragma unroll
      for (int r = 0; r < 8; ++r)
        sPw[(r + 8 * hlf) * 64 + t * 16 + cn] = f2bf_rne(s[t][r]);

    // O += P @ V   (K dim = 64 keys -> two K=32 blocks)
#pragma unroll
    for (int kk = 0; kk < 2; ++kk) {
      v16bf ap = load_frag_a(sPw + kk * 32, 64);
#pragma unroll
      for (int t = 0; t < 4; ++t) {
        v16bf bv = load_frag_b(sVt + (t * 16) * 64 + kk * 32, 64);
        o[t] = wmma_bf16(ap, bv, o[t]);
      }
    }
  }

  // epilogue: normalize and store bf16
#pragma unroll
  for (int r = 0; r < 8; ++r) li[r] = 1.0f / li[r];
#pragma unroll
  for (int t = 0; t < 4; ++t)
#pragma unroll
    for (int r = 0; r < 8; ++r) {
      size_t off = (size_t)(row0 + r + 8 * hlf) * DMODEL + h * DHEAD + t * 16 + cn;
      O[off] = f2bf_rne(o[t][r] * li[r]);
    }
}

// ---------------------------------------------------------------- launch

extern "C" void kernel_launch(void* const* d_in, const int* in_sizes, int n_in,
                              void* d_out, int out_size, void* d_ws, size_t ws_size,
                              hipStream_t stream) {
  (void)in_sizes; (void)n_in; (void)out_size; (void)ws_size;
  const float* q  = (const float*)d_in[0];
  const float* k  = (const float*)d_in[1];
  const float* v  = (const float*)d_in[2];
  // d_in[3] is the causal mask; causality is computed analytically.
  const float* Wq = (const float*)d_in[4];
  const float* Wk = (const float*)d_in[5];
  const float* Wv = (const float*)d_in[6];
  const float* Wo = (const float*)d_in[7];

  const size_t NX = (size_t)SEQLEN * DMODEL;   // 4M elems
  const size_t NW = (size_t)DMODEL * DMODEL;   // 1M elems
  u16* ws = (u16*)d_ws;                        // ~64 MiB of scratch used
  u16* xq = ws;
  u16* xk = xq + NX;
  u16* xv = xk + NX;
  u16* wq = xv + NX;
  u16* wk = wq + NW;
  u16* wv = wk + NW;
  u16* wo = wv + NW;
  u16* Qb = wo + NW;
  u16* Kb = Qb + NX;
  u16* Vb = Kb + NX;
  u16* Ob = Vb + NX;

  const int T = 256;
  f32_to_bf16_kernel<<<(int)((NX + T - 1) / T), T, 0, stream>>>(q, xq, (int)NX);
  f32_to_bf16_kernel<<<(int)((NX + T - 1) / T), T, 0, stream>>>(k, xk, (int)NX);
  f32_to_bf16_kernel<<<(int)((NX + T - 1) / T), T, 0, stream>>>(v, xv, (int)NX);
  f32_to_bf16_kernel<<<(int)((NW + T - 1) / T), T, 0, stream>>>(Wq, wq, (int)NW);
  f32_to_bf16_kernel<<<(int)((NW + T - 1) / T), T, 0, stream>>>(Wk, wk, (int)NW);
  f32_to_bf16_kernel<<<(int)((NW + T - 1) / T), T, 0, stream>>>(Wv, wv, (int)NW);
  f32_to_bf16_kernel<<<(int)((NW + T - 1) / T), T, 0, stream>>>(Wo, wo, (int)NW);

  dim3 gg(SEQLEN / 64, DMODEL / 64);   // (64, 16)
  dim3 gb(128);
  gemm_abt<false><<<gg, gb, 0, stream>>>(xq, wq, Qb, SEQLEN, DMODEL, DMODEL);
  gemm_abt<false><<<gg, gb, 0, stream>>>(xk, wk, Kb, SEQLEN, DMODEL, DMODEL);
  gemm_abt<false><<<gg, gb, 0, stream>>>(xv, wv, Vb, SEQLEN, DMODEL, DMODEL);

  flash_attn_kernel<<<dim3(SEQLEN / 64, NHEADS), dim3(128), 0, stream>>>(Qb, Kb, Vb, Ob);

  gemm_abt<true><<<gg, gb, 0, stream>>>(Ob, wo, d_out, SEQLEN, DMODEL, DMODEL);
}
